// PeepholeLSTMCell_19619410608300
// MI455X (gfx1250) — matrix operand order repaired and checked by reading, not compiled, not run on hardware
//
#include <hip/hip_runtime.h>
#include <hip/hip_bf16.h>
#include <hip/hip_fp16.h>

typedef __attribute__((ext_vector_type(16))) _Float16 v16h;
typedef __attribute__((ext_vector_type(8)))  _Float16 v8h;
typedef __attribute__((ext_vector_type(4)))  _Float16 v4h;
typedef __attribute__((ext_vector_type(8)))  float    v8f;

static constexpr int Bsz = 256;   // batch
static constexpr int Tsz = 512;   // time
static constexpr int Dsz = 256;   // input dim
static constexpr int Usz = 512;   // hidden dim

union AFrag { v16h v; v8h h[2]; };

// ---------------------------------------------------------------------------
// Prep kernels (run once per launch; packed layouts make WMMA fragment loads
// contiguous 16-byte global_load_b128)
// ---------------------------------------------------------------------------

__global__ void cast_x_f16(const float* __restrict__ x, _Float16* __restrict__ x16, int n4) {
    int i = blockIdx.x * blockDim.x + threadIdx.x;
    if (i < n4) {
        float4 v = reinterpret_cast<const float4*>(x)[i];
        v4h o = { (_Float16)v.x, (_Float16)v.y, (_Float16)v.z, (_Float16)v.w };
        reinterpret_cast<v4h*>(x16)[i] = o;
    }
}

// WT[(g*U + u)*D + d] = W_g[d*U + u]   (B-matrix stored K-contiguous per column)
__global__ void pack_W(const float* __restrict__ Wf, const float* __restrict__ Wi,
                       const float* __restrict__ Wc, const float* __restrict__ Wo,
                       _Float16* __restrict__ WT) {
    int idx = blockIdx.x * blockDim.x + threadIdx.x;
    if (idx >= 4 * Usz * Dsz) return;
    int gr = idx / Dsz, d = idx - gr * Dsz;
    int g = gr / Usz, u = gr - g * Usz;
    const float* W = (g == 0) ? Wf : (g == 1) ? Wi : (g == 2) ? Wc : Wo;
    WT[idx] = (_Float16)W[d * Usz + u];
}

// UT[(g*U + u)*U + j] = U_g[j*U + u]
__global__ void pack_U(const float* __restrict__ Uf, const float* __restrict__ Ui,
                       const float* __restrict__ Uc, const float* __restrict__ Uo,
                       _Float16* __restrict__ UT) {
    int idx = blockIdx.x * blockDim.x + threadIdx.x;
    if (idx >= 4 * Usz * Usz) return;
    int gr = idx / Usz, j = idx - gr * Usz;
    int g = gr / Usz, u = gr - g * Usz;
    const float* Ur = (g == 0) ? Uf : (g == 1) ? Ui : (g == 2) ? Uc : Uo;
    UT[idx] = (_Float16)Ur[j * Usz + u];
}

__global__ void zero_state(_Float16* __restrict__ h0, float* __restrict__ c) {
    int idx = blockIdx.x * blockDim.x + threadIdx.x;
    if (idx < Bsz * Usz) { h0[idx] = (_Float16)0.0f; c[idx] = 0.0f; }
}

// ---------------------------------------------------------------------------
// One LSTM timestep: z = [x_t, h] @ [W; Ur]  (K = D + U = 768), fused gates.
// Wave tile: M=32 rows (batch) x N=16 cols (u) x 4 gates -> 8 WMMAs / K-step.
// Block: 128 threads = 4 waves spanning 64 u-columns; grid (U/64, B/32).
// ---------------------------------------------------------------------------
__global__ __launch_bounds__(128)
void lstm_step(const _Float16* __restrict__ x16,   // [B, T, D] f16
               const _Float16* __restrict__ WT,    // [4U, D]  f16 (Bᵀ packed)
               const _Float16* __restrict__ UT,    // [4U, U]  f16 (Bᵀ packed)
               const _Float16* __restrict__ hprev, // [B, U]   f16
               _Float16* __restrict__ hnext,       // [B, U]   f16
               float* __restrict__ cbuf,           // [B, U]   f32 cell state
               const float* __restrict__ Pf, const float* __restrict__ bf,
               const float* __restrict__ Pi, const float* __restrict__ bi,
               const float* __restrict__ bc,
               const float* __restrict__ Po, const float* __restrict__ bo,
               float* __restrict__ out,            // [B, T, U] f32
               int t)
{
    const int lane  = threadIdx.x & 31;
    const int wave  = threadIdx.x >> 5;               // 0..3 (column waves)
    const int n0    = blockIdx.x * 64 + wave * 16;    // u tile base
    const int m0    = blockIdx.y * 32;                // batch tile base
    const int lhalf = (lane < 16) ? 0 : 1;
    const int l15   = lane & 15;

    // A-fragment addressing (16-bit A 16x32 layout):
    // lanes 0-15: row m0+l, K chunks [k..k+7] and [k+16..k+23]
    // lanes16-31: row m0+l, K chunks [k+8..k+15] and [k+24..k+31]
    const int rowA0 = m0 + l15;
    const int rowA1 = m0 + 16 + l15;
    const int kA0   = lhalf * 8;

    // B-fragment addressing: lane column = n, K chunks same as A ([k+8h..+7] and [k+16+8h..+7])
    const int colN = n0 + l15;
    const int kB   = lhalf * 8;

    const _Float16* xrow0 = x16 + ((size_t)rowA0 * Tsz + t) * Dsz;
    const _Float16* xrow1 = x16 + ((size_t)rowA1 * Tsz + t) * Dsz;
    const _Float16* hrow0 = hprev + (size_t)rowA0 * Usz;
    const _Float16* hrow1 = hprev + (size_t)rowA1 * Usz;

    v8f vzero = {};
    v8f acc[2][4];
    #pragma unroll
    for (int i = 0; i < 2; ++i)
        #pragma unroll
        for (int g = 0; g < 4; ++g) acc[i][g] = vzero;

    // ---- K over D: x_t @ W ----
    for (int k = 0; k < Dsz; k += 32) {
        AFrag a0, a1;
        a0.h[0] = *reinterpret_cast<const v8h*>(xrow0 + k + kA0);
        a0.h[1] = *reinterpret_cast<const v8h*>(xrow0 + k + kA0 + 16);
        a1.h[0] = *reinterpret_cast<const v8h*>(xrow1 + k + kA0);
        a1.h[1] = *reinterpret_cast<const v8h*>(xrow1 + k + kA0 + 16);
        #pragma unroll
        for (int g = 0; g < 4; ++g) {
            const _Float16* bp = WT + (size_t)(g * Usz + colN) * Dsz + k + kB;
            AFrag b;
            b.h[0] = *reinterpret_cast<const v8h*>(bp);
            b.h[1] = *reinterpret_cast<const v8h*>(bp + 16);
            acc[0][g] = __builtin_amdgcn_wmma_f32_16x16x32_f16(
                false, a0.v, false, b.v, (short)0, acc[0][g], false, false);
            acc[1][g] = __builtin_amdgcn_wmma_f32_16x16x32_f16(
                false, a1.v, false, b.v, (short)0, acc[1][g], false, false);
        }
    }

    // ---- K over U: h @ Ur ----
    for (int k = 0; k < Usz; k += 32) {
        AFrag a0, a1;
        a0.h[0] = *reinterpret_cast<const v8h*>(hrow0 + k + kA0);
        a0.h[1] = *reinterpret_cast<const v8h*>(hrow0 + k + kA0 + 16);
        a1.h[0] = *reinterpret_cast<const v8h*>(hrow1 + k + kA0);
        a1.h[1] = *reinterpret_cast<const v8h*>(hrow1 + k + kA0 + 16);
        #pragma unroll
        for (int g = 0; g < 4; ++g) {
            const _Float16* bp = UT + (size_t)(g * Usz + colN) * Usz + k + kB;
            AFrag b;
            b.h[0] = *reinterpret_cast<const v8h*>(bp);
            b.h[1] = *reinterpret_cast<const v8h*>(bp + 16);
            acc[0][g] = __builtin_amdgcn_wmma_f32_16x16x32_f16(
                false, a0.v, false, b.v, (short)0, acc[0][g], false, false);
            acc[1][g] = __builtin_amdgcn_wmma_f32_16x16x32_f16(
                false, a1.v, false, b.v, (short)0, acc[1][g], false, false);
        }
    }

    // ---- Fused gate epilogue (f32) ----
    const int u = colN;
    const float pf = Pf[u], vbf = bf[u];
    const float pi = Pi[u], vbi = bi[u];
    const float vbc = bc[u];
    const float po = Po[u], vbo = bo[u];

    #pragma unroll
    for (int tile = 0; tile < 2; ++tile) {
        #pragma unroll
        for (int r = 0; r < 8; ++r) {
            // C/D layout: lanes 0-15 -> M = r, lanes 16-31 -> M = r + 8
            const int m = m0 + tile * 16 + lhalf * 8 + r;
            const size_t cidx = (size_t)m * Usz + u;
            const float cold = cbuf[cidx];
            const float zf = acc[tile][0][r];
            const float zi = acc[tile][1][r];
            const float zc = acc[tile][2][r];
            const float zo = acc[tile][3][r];
            const float fg = 1.0f / (1.0f + __expf(-(zf + cold * pf + vbf)));
            const float ig = 1.0f / (1.0f + __expf(-(zi + cold * pi + vbi)));
            const float cc = tanhf(zc + vbc);
            const float cn = fg * cold + ig * cc;
            const float og = 1.0f / (1.0f + __expf(-(zo + cn * po + vbo)));
            const float hn = og * tanhf(cn);
            cbuf[cidx] = cn;
            hnext[cidx] = (_Float16)hn;
            out[((size_t)m * Tsz + t) * Usz + u] = hn;
        }
    }
}

// ---------------------------------------------------------------------------
extern "C" void kernel_launch(void* const* d_in, const int* in_sizes, int n_in,
                              void* d_out, int out_size, void* d_ws, size_t ws_size,
                              hipStream_t stream) {
    const float* x   = (const float*)d_in[0];
    const float* W_f = (const float*)d_in[1];
    const float* U_f = (const float*)d_in[2];
    const float* P_f = (const float*)d_in[3];
    const float* b_f = (const float*)d_in[4];
    const float* W_i = (const float*)d_in[5];
    const float* U_i = (const float*)d_in[6];
    const float* P_i = (const float*)d_in[7];
    const float* b_i = (const float*)d_in[8];
    const float* W_c = (const float*)d_in[9];
    const float* U_c = (const float*)d_in[10];
    const float* b_c = (const float*)d_in[11];
    const float* W_o = (const float*)d_in[12];
    const float* U_o = (const float*)d_in[13];
    const float* P_o = (const float*)d_in[14];
    const float* b_o = (const float*)d_in[15];

    // Workspace layout (all chunks 256B-aligned by construction)
    char* ws = (char*)d_ws;
    _Float16* x16 = (_Float16*)ws;  ws += (size_t)Bsz * Tsz * Dsz * sizeof(_Float16); // 64 MB
    _Float16* WT  = (_Float16*)ws;  ws += (size_t)4 * Usz * Dsz * sizeof(_Float16);   // 1 MB
    _Float16* UT  = (_Float16*)ws;  ws += (size_t)4 * Usz * Usz * sizeof(_Float16);   // 2 MB
    _Float16* h0  = (_Float16*)ws;  ws += (size_t)Bsz * Usz * sizeof(_Float16);
    _Float16* h1  = (_Float16*)ws;  ws += (size_t)Bsz * Usz * sizeof(_Float16);
    float*    cst = (float*)ws;     ws += (size_t)Bsz * Usz * sizeof(float);

    // Prep (re-run every call: deterministic, re-zeroes mutated state)
    {
        int n4 = (Bsz * Tsz * Dsz) / 4;
        cast_x_f16<<<(n4 + 255) / 256, 256, 0, stream>>>(x, x16, n4);
        int nW = 4 * Usz * Dsz;
        pack_W<<<(nW + 255) / 256, 256, 0, stream>>>(W_f, W_i, W_c, W_o, WT);
        int nU = 4 * Usz * Usz;
        pack_U<<<(nU + 255) / 256, 256, 0, stream>>>(U_f, U_i, U_c, U_o, UT);
        int nS = Bsz * Usz;
        zero_state<<<(nS + 255) / 256, 256, 0, stream>>>(h0, cst);
    }

    // Sequential recurrence: one kernel per timestep, double-buffered h
    dim3 grid(Usz / 64, Bsz / 32);
    for (int t = 0; t < Tsz; ++t) {
        _Float16* hp = (t & 1) ? h1 : h0;
        _Float16* hn = (t & 1) ? h0 : h1;
        lstm_step<<<grid, 128, 0, stream>>>(x16, WT, UT, hp, hn, cst,
                                            P_f, b_f, P_i, b_i, b_c, P_o, b_o,
                                            (float*)d_out, t);
    }
}
